// QuantQwenAttentionV2_88862873354593
// MI455X (gfx1250) — 16-bit and 8-bit matrix operand order repaired and checked by reading, compiled, not run on hardware
//
#include <hip/hip_runtime.h>
#include <hip/hip_bf16.h>

typedef __attribute__((ext_vector_type(8)))  int       v8i;
typedef __attribute__((ext_vector_type(8)))  float     v8f;
typedef __attribute__((ext_vector_type(16))) _Float16  v16h;
typedef __attribute__((ext_vector_type(8)))  _Float16  v8h;
typedef signed char i8;

#define S_LEN 2048
#define HID   3584
#define NQ    3584
#define NKV   512
#define NHEAD 28
#define NKVH  4
#define HD    128
#define GQA   7


static __device__ __forceinline__ v8i load_afrag_i8(const i8* base, int stride, int lane) {
  const i8* p = base + (size_t)(lane & 15) * stride + ((lane >> 4) << 3);
  const int2 t0 = *(const int2*)(p);
  const int2 t1 = *(const int2*)(p + 16);
  const int2 t2 = *(const int2*)(p + 32);
  const int2 t3 = *(const int2*)(p + 48);
  v8i f;
  f[0] = t0.x; f[1] = t0.y; f[2] = t1.x; f[3] = t1.y;
  f[4] = t2.x; f[5] = t2.y; f[6] = t3.x; f[7] = t3.y;
  return f;
}

static __device__ __forceinline__ v8i load_bfrag_i8(const i8* base, int stride, int lane) {
  const i8* p = base + (size_t)(lane & 15) * stride + ((lane >> 4) << 3);
  const int2 t0 = *(const int2*)(p);
  const int2 t1 = *(const int2*)(p + 16);
  const int2 t2 = *(const int2*)(p + 32);
  const int2 t3 = *(const int2*)(p + 48);
  v8i f;
  f[0] = t0.x; f[1] = t0.y; f[2] = t1.x; f[3] = t1.y;
  f[4] = t2.x; f[5] = t2.y; f[6] = t3.x; f[7] = t3.y;
  return f;
}

static __device__ __forceinline__ v16h load_afrag_f16(const _Float16* base, int stride, int lane) {
  const _Float16* p = base + (size_t)(lane & 15) * stride + ((lane >> 4) << 3);
  v8h lo = *(const v8h*)(p);
  v8h hi = *(const v8h*)(p + 16);
  v16h f;
#pragma unroll
  for (int i = 0; i < 8; ++i) { f[i] = lo[i]; f[8 + i] = hi[i]; }
  return f;
}

static __device__ __forceinline__ v16h load_bfrag_f16(const _Float16* base, int stride, int lane) {
  const _Float16* p = base + (size_t)(lane & 15) * stride + ((lane >> 4) << 3);
  v8h lo = *(const v8h*)(p);
  v8h hi = *(const v8h*)(p + 16);
  v16h f;
#pragma unroll
  for (int i = 0; i < 8; ++i) { f[i] = lo[i]; f[8 + i] = hi[i]; }
  return f;
}


__global__ __launch_bounds__(32) void zero_amax_kernel(unsigned* amax) {
  if (threadIdx.x < 2) amax[threadIdx.x] = 0u;
}

__global__ __launch_bounds__(256) void amax_kernel(const float* __restrict__ x, int n,
                                                   unsigned* __restrict__ out) {
  float m = 0.0f;
  for (int i = blockIdx.x * blockDim.x + threadIdx.x; i < n; i += blockDim.x * gridDim.x)
    m = fmaxf(m, fabsf(x[i]));
#pragma unroll
  for (int off = 16; off > 0; off >>= 1) m = fmaxf(m, __shfl_xor(m, off));
  __shared__ float red[8];
  if ((threadIdx.x & 31) == 0) red[threadIdx.x >> 5] = m;
  __syncthreads();
  if (threadIdx.x == 0) {
    float mm = red[0];
#pragma unroll
    for (int i = 1; i < 8; ++i) mm = fmaxf(mm, red[i]);
    atomicMax(out, __float_as_uint(mm));
  }
}

__global__ __launch_bounds__(256) void quant_act_kernel(const float* __restrict__ x,
                                                        i8* __restrict__ xq,
                                                        const unsigned* __restrict__ amax, int n) {
  const float s = fmaxf(__uint_as_float(*amax) * (1.0f / 127.0f), 1e-8f);
  const float inv = 1.0f / s;
  for (int i = blockIdx.x * blockDim.x + threadIdx.x; i < n; i += blockDim.x * gridDim.x) {
    float q = rintf(x[i] * inv);
    q = fminf(fmaxf(q, -128.0f), 127.0f);
    xq[i] = (i8)q;
  }
}

__global__ __launch_bounds__(256) void quant_weight_kernel(const float* __restrict__ w,
                                                           i8* __restrict__ wq,
                                                           float* __restrict__ wscale) {
  const int row = blockIdx.x;
  const float* wr = w + (size_t)row * HID;
  float m = 0.0f;
  for (int i = threadIdx.x; i < HID; i += 256) m = fmaxf(m, fabsf(wr[i]));
  __shared__ float red[256];
  red[threadIdx.x] = m;
  __syncthreads();
  for (int st = 128; st > 0; st >>= 1) {
    if (threadIdx.x < st) red[threadIdx.x] = fmaxf(red[threadIdx.x], red[threadIdx.x + st]);
    __syncthreads();
  }
  const float sc = fmaxf(red[0] * (1.0f / 7.0f), 1e-8f);
  if (threadIdx.x == 0) wscale[row] = sc;
  const float inv = 1.0f / sc;
  for (int i = threadIdx.x; i < HID; i += 256) {
    float q = rintf(wr[i] * inv);
    q = fminf(fmaxf(q, -8.0f), 7.0f);
    wq[(size_t)row * HID + i] = (i8)q;
  }
}

__global__ __launch_bounds__(256) void gemm_i8_kernel(const i8* __restrict__ A,
                                                      const i8* __restrict__ W,
                                                      const float* __restrict__ wscale,
                                                      const unsigned* __restrict__ amax_bits,
                                                      const float* __restrict__ bias,
                                                      float* __restrict__ out, int N) {
  const int lane = threadIdx.x & 31;
  const int wave = threadIdx.x >> 5;
  const int m0 = blockIdx.y * 128 + (wave >> 1) * 32;
  const int n0 = blockIdx.x * 64 + (wave & 1) * 32;
  v8i acc[2][2];
  {
    v8i z = {};
    acc[0][0] = z; acc[0][1] = z; acc[1][0] = z; acc[1][1] = z;
  }
  const i8* a0 = A + (size_t)m0 * HID;
  const i8* a1 = a0 + (size_t)16 * HID;
  const i8* b0 = W + (size_t)n0 * HID;
  const i8* b1 = b0 + (size_t)16 * HID;
#pragma unroll 4
  for (int k0 = 0; k0 < HID; k0 += 64) {
    __builtin_prefetch(a0 + k0 + 256, 0, 3);
    __builtin_prefetch(a1 + k0 + 256, 0, 3);
    __builtin_prefetch(b0 + k0 + 256, 0, 3);
    __builtin_prefetch(b1 + k0 + 256, 0, 3);
    v8i af0 = load_afrag_i8(a0 + k0, HID, lane);
    v8i af1 = load_afrag_i8(a1 + k0, HID, lane);
    v8i bf0 = load_bfrag_i8(b0 + k0, HID, lane);
    v8i bf1 = load_bfrag_i8(b1 + k0, HID, lane);
    acc[0][0] = __builtin_amdgcn_wmma_i32_16x16x64_iu8(true, af0, true, bf0, acc[0][0], false, false);
    acc[0][1] = __builtin_amdgcn_wmma_i32_16x16x64_iu8(true, af0, true, bf1, acc[0][1], false, false);
    acc[1][0] = __builtin_amdgcn_wmma_i32_16x16x64_iu8(true, af1, true, bf0, acc[1][0], false, false);
    acc[1][1] = __builtin_amdgcn_wmma_i32_16x16x64_iu8(true, af1, true, bf1, acc[1][1], false, false);
  }
  const float sa = fmaxf(__uint_as_float(*amax_bits) * (1.0f / 127.0f), 1e-8f);
#pragma unroll
  for (int tm = 0; tm < 2; ++tm) {
#pragma unroll
    for (int tn = 0; tn < 2; ++tn) {
      const int n = n0 + tn * 16 + (lane & 15);
      const int mb = m0 + tm * 16 + ((lane >> 4) << 3);
      const float sc = sa * wscale[n];
      const float bz = bias ? bias[n] : 0.0f;
#pragma unroll
      for (int r = 0; r < 8; ++r)
        out[(size_t)(mb + r) * N + n] = (float)acc[tm][tn][r] * sc + bz;
    }
  }
}

__global__ __launch_bounds__(128) void rope_convert_kernel(const float* __restrict__ qf,
                                                           const float* __restrict__ kf,
                                                           const float* __restrict__ vf,
                                                           const float* __restrict__ cosp,
                                                           const float* __restrict__ sinp,
                                                           _Float16* __restrict__ q16,
                                                           _Float16* __restrict__ k16,
                                                           _Float16* __restrict__ vT16) {
  const int s = blockIdx.x;
  const int job = blockIdx.y;
  const int d = threadIdx.x;
  if (job >= NHEAD + NKVH) {
    const int kv = job - (NHEAD + NKVH);
    const float v = vf[(size_t)s * NKV + kv * HD + d];
    vT16[((size_t)kv * HD + d) * S_LEN + s] = (_Float16)v;
    return;
  }
  const int axis = (d < 32) ? 0 : (d < 80) ? 1 : 2;
  const float c  = cosp[((size_t)axis * S_LEN + s) * HD + d];
  const float si = sinp[((size_t)axis * S_LEN + s) * HD + d];
  if (job < NHEAD) {
    const float* row = qf + (size_t)s * NQ + job * HD;
    const float x  = row[d];
    const float xo = (d < 64) ? -row[d + 64] : row[d - 64];
    q16[((size_t)job * S_LEN + s) * HD + d] = (_Float16)(x * c + xo * si);
  } else {
    const int kv = job - NHEAD;
    const float* row = kf + (size_t)s * NKV + kv * HD;
    const float x  = row[d];
    const float xo = (d < 64) ? -row[d + 64] : row[d - 64];
    k16[((size_t)kv * S_LEN + s) * HD + d] = (_Float16)(x * c + xo * si);
  }
}

__global__ __launch_bounds__(256) void flash_attn_kernel(const _Float16* __restrict__ q16,
                                                         const _Float16* __restrict__ k16,
                                                         const _Float16* __restrict__ vT16,
                                                         float* __restrict__ attnf) {
  __shared__ __align__(16) _Float16 pshm[8][16 * 32];
  const int lane = threadIdx.x & 31;
  const int wave = threadIdx.x >> 5;
  const int qb = blockIdx.x;
  const int h = blockIdx.y;
  const int kv = h / GQA;
  const int q0 = qb * 128 + wave * 16;
  const float scale = 0.08838834764831845f;
  const float NEG_INF = -__builtin_inff();

  v16h qfrag[4];
#pragma unroll
  for (int c = 0; c < 4; ++c)
    qfrag[c] = load_afrag_f16(q16 + ((size_t)h * S_LEN + q0) * HD + c * 32, HD, lane);

  v8f zero = {};
  float mrow[8], lrow[8];
  v8f oacc[8];
#pragma unroll
  for (int r = 0; r < 8; ++r) { mrow[r] = NEG_INF; lrow[r] = 0.0f; }
#pragma unroll
  for (int t = 0; t < 8; ++t) oacc[t] = zero;

  const int n = lane & 15;
  const int mb = (lane >> 4) << 3;
  const int nkb = (q0 + 47) >> 5;
  const _Float16* kbase = k16 + (size_t)kv * S_LEN * HD;
  const _Float16* vbase = vT16 + (size_t)kv * HD * S_LEN;

  for (int j = 0; j < nkb; ++j) {
    __builtin_prefetch(kbase + (size_t)(j * 32 + 32) * HD + (size_t)(lane & 15) * HD, 0, 3);
    v8f s0 = zero, s1 = zero;
#pragma unroll
    for (int c = 0; c < 4; ++c) {
      v16h b0 = load_bfrag_f16(kbase + (size_t)(j * 32) * HD + c * 32, HD, lane);
      s0 = __builtin_amdgcn_wmma_f32_16x16x32_f16(false, qfrag[c], false, b0, (short)0, s0, false, false);
      v16h b1 = load_bfrag_f16(kbase + (size_t)(j * 32 + 16) * HD + c * 32, HD, lane);
      s1 = __builtin_amdgcn_wmma_f32_16x16x32_f16(false, qfrag[c], false, b1, (short)0, s1, false, false);
    }
    float a0[8], a1[8], rmax[8];
#pragma unroll
    for (int r = 0; r < 8; ++r) {
      const int m = q0 + mb + r;
      float x0 = s0[r] * scale; if (j * 32 + n > m)      x0 = NEG_INF;
      float x1 = s1[r] * scale; if (j * 32 + 16 + n > m) x1 = NEG_INF;
      a0[r] = x0; a1[r] = x1;
      float rm = fmaxf(x0, x1);
#pragma unroll
      for (int off = 1; off < 16; off <<= 1) rm = fmaxf(rm, __shfl_xor(rm, off));
      rmax[r] = rm;
    }
    float alpha[8];
#pragma unroll
    for (int r = 0; r < 8; ++r) {
      const float mnew = fmaxf(mrow[r], rmax[r]);
      alpha[r] = __expf(mrow[r] - mnew);
      mrow[r] = mnew;
      const float p0 = __expf(a0[r] - mnew);
      const float p1 = __expf(a1[r] - mnew);
      a0[r] = p0; a1[r] = p1;
      float rs = p0 + p1;
#pragma unroll
      for (int off = 1; off < 16; off <<= 1) rs += __shfl_xor(rs, off);
      lrow[r] = lrow[r] * alpha[r] + rs;
    }
#pragma unroll
    for (int t = 0; t < 8; ++t)
#pragma unroll
      for (int r = 0; r < 8; ++r) oacc[t][r] *= alpha[r];
    _Float16* pw = &pshm[wave][0];
#pragma unroll
    for (int r = 0; r < 8; ++r) {
      pw[(mb + r) * 32 + n]      = (_Float16)a0[r];
      pw[(mb + r) * 32 + n + 16] = (_Float16)a1[r];
    }
    asm volatile("s_wait_dscnt 0" ::: "memory");
    v16h pa = load_afrag_f16(pw, 32, lane);
#pragma unroll
    for (int t = 0; t < 8; ++t) {
      v16h vb = load_bfrag_f16(vbase + (size_t)(t * 16) * S_LEN + j * 32, S_LEN, lane);
      oacc[t] = __builtin_amdgcn_wmma_f32_16x16x32_f16(false, pa, false, vb, (short)0, oacc[t], false, false);
    }
  }
#pragma unroll
  for (int t = 0; t < 8; ++t) {
#pragma unroll
    for (int r = 0; r < 8; ++r) {
      const int m = q0 + mb + r;
      attnf[(size_t)m * HID + h * HD + t * 16 + n] = oacc[t][r] / lrow[r];
    }
  }
}

extern "C" void kernel_launch(void* const* d_in, const int* in_sizes, int n_in,
                              void* d_out, int out_size, void* d_ws, size_t ws_size,
                              hipStream_t stream) {
  const float* hidden = (const float*)d_in[0];
  const float* cosp   = (const float*)d_in[1];
  const float* sinp   = (const float*)d_in[2];
  const float* wqw    = (const float*)d_in[3];
  const float* bq     = (const float*)d_in[4];
  const float* wkw    = (const float*)d_in[5];
  const float* bk     = (const float*)d_in[6];
  const float* wvw    = (const float*)d_in[7];
  const float* bv     = (const float*)d_in[8];
  const float* wow    = (const float*)d_in[9];
  float* outp = (float*)d_out;
  (void)in_sizes; (void)n_in; (void)out_size; (void)ws_size;

  char* base = (char*)d_ws;
  size_t cur = 0;
  auto alloc = [&](size_t bytes) -> void* {
    cur = (cur + 255) & ~(size_t)255;
    void* p = base + cur;
    cur += bytes;
    return p;
  };
  unsigned* amax = (unsigned*)alloc(2 * sizeof(unsigned));
  float* wq_s = (float*)alloc((size_t)NQ * sizeof(float));
  float* wk_s = (float*)alloc((size_t)NKV * sizeof(float));
  float* wv_s = (float*)alloc((size_t)NKV * sizeof(float));
  float* wo_s = (float*)alloc((size_t)NQ * sizeof(float));
  i8* xq  = (i8*)alloc((size_t)S_LEN * HID);
  i8* wqq = (i8*)alloc((size_t)NQ * HID);
  i8* wkq = (i8*)alloc((size_t)NKV * HID);
  i8* wvq = (i8*)alloc((size_t)NKV * HID);
  i8* woq = (i8*)alloc((size_t)NQ * HID);
  float* qf = (float*)alloc((size_t)S_LEN * NQ * sizeof(float));
  float* kf = (float*)alloc((size_t)S_LEN * NKV * sizeof(float));
  float* vf = (float*)alloc((size_t)S_LEN * NKV * sizeof(float));
  _Float16* q16  = (_Float16*)alloc((size_t)NHEAD * S_LEN * HD * 2);
  _Float16* k16  = (_Float16*)alloc((size_t)NKVH * S_LEN * HD * 2);
  _Float16* vT16 = (_Float16*)alloc((size_t)NKVH * HD * S_LEN * 2);

  zero_amax_kernel<<<1, 32, 0, stream>>>(amax);
  amax_kernel<<<512, 256, 0, stream>>>(hidden, S_LEN * HID, amax + 0);
  quant_act_kernel<<<512, 256, 0, stream>>>(hidden, xq, amax + 0, S_LEN * HID);
  quant_weight_kernel<<<NQ, 256, 0, stream>>>(wqw, wqq, wq_s);
  quant_weight_kernel<<<NKV, 256, 0, stream>>>(wkw, wkq, wk_s);
  quant_weight_kernel<<<NKV, 256, 0, stream>>>(wvw, wvq, wv_s);
  quant_weight_kernel<<<NQ, 256, 0, stream>>>(wow, woq, wo_s);

  gemm_i8_kernel<<<dim3(NQ / 64, S_LEN / 128), 256, 0, stream>>>(xq, wqq, wq_s, amax + 0, bq, qf, NQ);
  gemm_i8_kernel<<<dim3(NKV / 64, S_LEN / 128), 256, 0, stream>>>(xq, wkq, wk_s, amax + 0, bk, kf, NKV);
  gemm_i8_kernel<<<dim3(NKV / 64, S_LEN / 128), 256, 0, stream>>>(xq, wvq, wv_s, amax + 0, bv, vf, NKV);

  rope_convert_kernel<<<dim3(S_LEN, NHEAD + 2 * NKVH), 128, 0, stream>>>(
      qf, kf, vf, cosp, sinp, q16, k16, vT16);

  float* attnf = qf;
  flash_attn_kernel<<<dim3(S_LEN / 128, NHEAD), 256, 0, stream>>>(q16, k16, vT16, attnf);

  amax_kernel<<<512, 256, 0, stream>>>(attnf, S_LEN * HID, amax + 1);
  i8* attnq = xq;
  quant_act_kernel<<<512, 256, 0, stream>>>(attnf, attnq, amax + 1, S_LEN * HID);
  gemm_i8_kernel<<<dim3(NQ / 64, S_LEN / 128), 256, 0, stream>>>(
      attnq, woq, wo_s, amax + 1, nullptr, outp, NQ);
}
